// Attention_6682969112611
// MI455X (gfx1250) — matrix operand order repaired and checked by reading, not compiled, not run on hardware
//
#include <hip/hip_runtime.h>

// ---------------------------------------------------------------------------
// Fused batch-softmax attention for MI455X (gfx1250, wave32, WMMA bf16)
//   B=16, C=2048, L=64
//   S = q @ reshape(k,(L,C)) * (1/8); softmax over batch axis; out = S @ v
//
// Pipeline:
//   (1) one-shot fp32->bf16 conversion of q (pre-scaled by 1/sqrt(L)), k, v
//       into d_ws  -> hot loop streams half the operand bytes, no per-tile cvt
//   (2) fused kernel: one WG = 16 waves, wave w owns batch b = w for a 16-row
//       m-tile. softmax(axis=0) is pointwise in (m,n) -> the 256MB score
//       tensor lives only in LDS/registers.
// Loop-level ILP: V fragments are issued at the top of each iteration so
// their latency hides under GEMM1 + barriers + softmax; all four K fragments
// are issued into distinct registers before the first WMMA so the scheduler
// can use partial s_wait_loadcnt values instead of full drains.
// ---------------------------------------------------------------------------

typedef __attribute__((ext_vector_type(16))) __bf16 v16bf;
typedef __attribute__((ext_vector_type(8)))  __bf16 v8bf;
typedef __attribute__((ext_vector_type(8)))  float  v8f;

#define BATCH 16
#define CDIM  2048
#define LDIM  64
#define CL    (CDIM * LDIM)
#define NELEM (BATCH * CL)          // 2,097,152 elements per tensor

// ---------------------------------------------------------------------------
// Fragment loaders: 16 consecutive K/V elements, or two 8-element runs (A).
// bf16 overloads: straight vector loads (global_load_b128 pairs).
// fp32 overloads: load + packed convert (fallback path only).
// ---------------------------------------------------------------------------
__device__ __forceinline__ v16bf frag16(const __bf16* p) {
    return *(const v16bf*)p;                       // 32B, 32B-aligned
}
__device__ __forceinline__ v16bf frag8x2(const __bf16* p0, const __bf16* p1) {
    v8bf lo = *(const v8bf*)p0;
    v8bf hi = *(const v8bf*)p1;
    v16bf r;
#pragma unroll
    for (int i = 0; i < 8; ++i) { r[i] = lo[i]; r[8 + i] = hi[i]; }
    return r;
}
__device__ __forceinline__ v16bf frag16(const float* p) {
    const float4* p4 = (const float4*)p;
    float4 x0 = p4[0], x1 = p4[1], x2 = p4[2], x3 = p4[3];
    float t[16] = {x0.x,x0.y,x0.z,x0.w, x1.x,x1.y,x1.z,x1.w,
                   x2.x,x2.y,x2.z,x2.w, x3.x,x3.y,x3.z,x3.w};
    v16bf r;
#pragma unroll
    for (int i = 0; i < 16; ++i) r[i] = (__bf16)t[i];
    return r;
}
__device__ __forceinline__ v16bf frag8x2(const float* p0, const float* p1) {
    const float4* a4 = (const float4*)p0;
    const float4* b4 = (const float4*)p1;
    float4 a0 = a4[0], a1 = a4[1], b0 = b4[0], b1 = b4[1];
    float t0[8] = {a0.x,a0.y,a0.z,a0.w, a1.x,a1.y,a1.z,a1.w};
    float t1[8] = {b0.x,b0.y,b0.z,b0.w, b1.x,b1.y,b1.z,b1.w};
    v16bf r;
#pragma unroll
    for (int i = 0; i < 8; ++i) { r[i] = (__bf16)t0[i]; r[8 + i] = (__bf16)t1[i]; }
    return r;
}

__device__ __forceinline__ v8f wmma_bf16(v16bf a, v16bf b, v8f c) {
    return __builtin_amdgcn_wmma_f32_16x16x32_bf16(
        /*neg_a=*/false, a, /*neg_b=*/false, b,
        /*c_mod=*/(short)0, c, /*reuse_a=*/false, /*reuse_b=*/false);
}

// ---------------------------------------------------------------------------
// Pre-pass: fp32 -> bf16 (optionally pre-scaled), 8 elements per thread
// ---------------------------------------------------------------------------
__global__ __launch_bounds__(256)
void cvt_bf16_kernel(const float* __restrict__ src, __bf16* __restrict__ dst,
                     float scale, int n) {
    int i = (blockIdx.x * 256 + threadIdx.x) * 8;
    if (i >= n) return;
    const float4* p4 = (const float4*)(src + i);
    float4 a = p4[0], b = p4[1];
    float t[8] = {a.x, a.y, a.z, a.w, b.x, b.y, b.z, b.w};
    v8bf r;
#pragma unroll
    for (int j = 0; j < 8; ++j) r[j] = (__bf16)(t[j] * scale);
    *(v8bf*)(dst + i) = r;
}

// ---------------------------------------------------------------------------
// Fused attention kernel. T = __bf16 (pre-converted, PRESCALED q) or float.
// ---------------------------------------------------------------------------
template <typename T, bool PRESCALED>
__global__ __launch_bounds__(512, 1)
void attn_fused_kernel(const T* __restrict__ q,
                       const T* __restrict__ k,
                       const T* __restrict__ v,
                       float* __restrict__ out)
{
    __shared__ float  Ssm[BATCH][16][32];   // scores, 32 KB
    __shared__ __bf16 Psm[BATCH][16][32];   // softmaxed probs, 16 KB

    const int m0   = blockIdx.x * 16;       // 128 m-tiles
    const int tid  = threadIdx.x;
    const int lane = tid & 31;              // wave32
    const int b    = tid >> 5;              // wave id == batch id
    const int M    = lane & 15;
    const int kh   = lane >> 4;

    const T* qb = q + b * CL + (m0 + M) * LDIM;
    const T* kb = k + b * CL;
    const T* vb = v + b * CL;

    // Q A-fragments (16x64 -> two K=32 fragments), loaded once.
    // A 16x32 layout: lane M=lane&15; elems 0..7 -> K = kh*8+{0..7},
    //                 elems 8..15 -> K = 16 + kh*8 + {0..7}
    v16bf aQ0 = frag8x2(qb +      kh * 8, qb + 16 + kh * 8);
    v16bf aQ1 = frag8x2(qb + 32 + kh * 8, qb + 48 + kh * 8);

    // out accumulators: 16x64 fp32 = four 16x16 C/D tiles (32 VGPRs)
    v8f acc0 = {}, acc1 = {}, acc2 = {}, acc3 = {};

    for (int n0 = 0; n0 < CDIM; n0 += 32) {
        // ---- issue V fragments FIRST: consumed only after the softmax, so
        //      their latency hides under GEMM1 + barriers + softmax ----------
        const T* vrow = vb + n0 * LDIM + M;   // lane = N column of V tile
        v16bf vf0, vf1, vf2, vf3;
#pragma unroll
        for (int s = 0; s < 16; ++s) {
            int kk = (s < 8) ? kh * 8 + s : 16 + kh * 8 + (s - 8);
            vf0[s] = (__bf16)vrow[kk * LDIM];
            vf1[s] = (__bf16)vrow[kk * LDIM + 16];
            vf2[s] = (__bf16)vrow[kk * LDIM + 32];
            vf3[s] = (__bf16)vrow[kk * LDIM + 48];
        }

        // ===== GEMM1: S[16x32] = Q[16x64] @ K_t[64x32] ======================
        // K_t[i][n] = k[b][i*32 + n/64][n%64]; per lane: column n0+M (+16),
        // elements over i in A's table K order.
        const int rbase = n0 >> 6;
        const int c0    = n0 & 63;
        const T* prow0 = kb + rbase * LDIM + c0 + M;
        const T* prow1 = kb + ((32 << 5) + rbase) * LDIM + c0 + M;
        // all four K fragments in flight before the first WMMA
        v16bf bk00, bk01, bk10, bk11;
#pragma unroll
        for (int s = 0; s < 16; ++s) {
            int kk = (s < 8) ? kh * 8 + s : 16 + kh * 8 + (s - 8);
            bk00[s] = (__bf16)prow0[(kk << 5) * LDIM];
            bk01[s] = (__bf16)prow0[(kk << 5) * LDIM + 16];
            bk10[s] = (__bf16)prow1[(kk << 5) * LDIM];
            bk11[s] = (__bf16)prow1[(kk << 5) * LDIM + 16];
        }

        v8f s0 = {}, s1 = {};
        s0 = wmma_bf16(aQ0, bk00, s0);
        s1 = wmma_bf16(aQ0, bk01, s1);
        s0 = wmma_bf16(aQ1, bk10, s0);
        s1 = wmma_bf16(aQ1, bk11, s1);

        // stash S in LDS (C/D layout: m = r + kh*8, n = M)
#pragma unroll
        for (int r = 0; r < 8; ++r) {
            int m = r + kh * 8;
            float e0 = s0[r], e1 = s1[r];
            if (!PRESCALED) { e0 *= 0.125f; e1 *= 0.125f; }  // 1/sqrt(L)
            Ssm[b][m][M]      = e0;
            Ssm[b][m][16 + M] = e1;
        }
        __syncthreads();

        // ===== softmax over the BATCH axis: pointwise in (m, nn) ============
        // wave b handles row m=b; lane handles nn=lane (conflict-free LDS)
        {
            const int m = b, nn = lane;
            float vals[BATCH];
            float mx = -3.0e38f;
#pragma unroll
            for (int bb = 0; bb < BATCH; ++bb) {
                vals[bb] = Ssm[bb][m][nn];
                mx = fmaxf(mx, vals[bb]);
            }
            float sum = 0.f;
#pragma unroll
            for (int bb = 0; bb < BATCH; ++bb) {
                vals[bb] = __expf(vals[bb] - mx);   // v_exp_f32
                sum += vals[bb];
            }
            float inv = 1.0f / sum;
#pragma unroll
            for (int bb = 0; bb < BATCH; ++bb)
                Psm[bb][m][nn] = (__bf16)(vals[bb] * inv);
        }
        __syncthreads();

        // ===== GEMM2: out[16x64] += P[16x32] @ V[32x64] =====================
        v16bf pf;
#pragma unroll
        for (int i = 0; i < 8; ++i) {
            pf[i]     = Psm[b][M][kh * 8 + i];
            pf[8 + i] = Psm[b][M][16 + kh * 8 + i];
        }
        acc0 = wmma_bf16(pf, vf0, acc0);
        acc1 = wmma_bf16(pf, vf1, acc1);
        acc2 = wmma_bf16(pf, vf2, acc2);
        acc3 = wmma_bf16(pf, vf3, acc3);

        // pull next v tile toward the caches (global_prefetch_b8)
        if (n0 + 32 < CDIM) __builtin_prefetch(vrow + 32 * LDIM, 0, 0);
    }

    // ===== store out[b, m0:m0+16, 0:64] =====================================
    float* ob = out + b * CL + m0 * LDIM;
#pragma unroll
    for (int r = 0; r < 8; ++r) {
        int m = r + kh * 8;
        ob[m * LDIM +      M] = acc0[r];
        ob[m * LDIM + 16 + M] = acc1[r];
        ob[m * LDIM + 32 + M] = acc2[r];
        ob[m * LDIM + 48 + M] = acc3[r];
    }
}

extern "C" void kernel_launch(void* const* d_in, const int* in_sizes, int n_in,
                              void* d_out, int out_size, void* d_ws, size_t ws_size,
                              hipStream_t stream) {
    const float* q = (const float*)d_in[0];
    const float* k = (const float*)d_in[1];
    const float* v = (const float*)d_in[2];
    float* out = (float*)d_out;

    const size_t need = (size_t)3 * NELEM * sizeof(__bf16);   // 12 MB
    dim3 agrid(CDIM / 16);   // 128 m-tiles
    dim3 ablock(512);        // 16 wave32 waves = 16 batches

    if (ws_size >= need) {
        __bf16* qh = (__bf16*)d_ws;
        __bf16* kh = qh + NELEM;
        __bf16* vh = kh + NELEM;
        dim3 cgrid(NELEM / (256 * 8)), cblock(256);
        // fold 1/sqrt(L) into q during conversion
        hipLaunchKernelGGL(cvt_bf16_kernel, cgrid, cblock, 0, stream, q, qh, 0.125f, NELEM);
        hipLaunchKernelGGL(cvt_bf16_kernel, cgrid, cblock, 0, stream, k, kh, 1.0f,   NELEM);
        hipLaunchKernelGGL(cvt_bf16_kernel, cgrid, cblock, 0, stream, v, vh, 1.0f,   NELEM);
        hipLaunchKernelGGL((attn_fused_kernel<__bf16, true>), agrid, ablock, 0, stream,
                           qh, kh, vh, out);
    } else {
        // workspace too small: fall back to on-the-fly conversion
        hipLaunchKernelGGL((attn_fused_kernel<float, false>), agrid, ablock, 0, stream,
                           q, k, v, out);
    }
}
